// CausalGraphLearner_26302379721023
// MI455X (gfx1250) — matrix operand order repaired and checked by reading, not compiled, not run on hardware
//
#include <hip/hip_runtime.h>
#include <hip/hip_bf16.h>

typedef __attribute__((ext_vector_type(16))) _Float16 v16h;
typedef __attribute__((ext_vector_type(8)))  float    v8f;

#define DD   128   // d
#define LL   32    // L
#define HH   32    // H
#define NTOK 256   // B*T

// -----------------------------------------------------------------------------
// Main fused kernel: one block per token n (256 blocks x 256 threads = 8 waves).
//  Phase 1: each wave computes a 16-row chunk of P = h*W1a^T and Q = h*W1b^T
//           with 4x v_wmma_f32_16x16x32_f16 (K=32 == L in ONE wmma), results
//           staged in LDS (row stride 33 floats -> conflict-free column reads).
//  Phase 2: ReLU-contraction over h with W2/b1 in registers; accumulate the
//           per-n score matrix into global S via float atomics.
// -----------------------------------------------------------------------------
__global__ __launch_bounds__(256)
void cgl_main(const float* __restrict__ z,    // (NTOK, 128, 32)
              const float* __restrict__ W1,   // (32, 64)  [W1a | W1b]
              const float* __restrict__ b1,   // (32)
              const float* __restrict__ W2,   // (32)
              float* __restrict__ S)          // (128,128) accumulator (pre-zeroed)
{
    __shared__ float Pl[DD * 33];
    __shared__ float Ql[DD * 33];

    const int n    = blockIdx.x;
    const int tid  = threadIdx.x;
    const int wave = tid >> 5;
    const int lane = tid & 31;
    const int ml   = lane & 15;   // M (A) / N (B,C,D) index within tile
    const int kh   = lane >> 4;   // K-half selector
    const int mb   = wave * 16;   // this wave's 16-row chunk of d

    // ---- A fragment (16x32 f16, ISA layout): two contiguous 8-float runs ----
    const float* hrow = z + ((size_t)n * DD + (mb + ml)) * LL + kh * 8;
    v16h a;
#pragma unroll
    for (int e = 0; e < 8; ++e) a[e]     = (_Float16)hrow[e];
#pragma unroll
    for (int e = 0; e < 8; ++e) a[e + 8] = (_Float16)hrow[16 + e];

    // ---- 4 WMMAs: (h-tile in {0,16}) x (pa: l_off=0 | pb: l_off=32) ----
#pragma unroll
    for (int t = 0; t < 4; ++t) {
        const int hb   = (t & 1) * 16;   // H tile base
        const int loff = (t >> 1) * 32;  // 0 -> W1a, 32 -> W1b
        float* dst = (t >> 1) ? Ql : Pl;

        // B fragment (32x16 f16): lane holds N = hb+ml, K = 8*kh + e (e<8), 16 + 8*kh + (e-8) (e>=8)
        const float* wrow = W1 + (hb + ml) * (2 * LL) + loff + kh * 8;
        v16h b;
#pragma unroll
        for (int e = 0; e < 16; ++e) b[e] = (_Float16)wrow[e + (e >= 8 ? 8 : 0)];

        v8f c = {};
        c = __builtin_amdgcn_wmma_f32_16x16x32_f16(
                /*neg_a=*/false, a, /*neg_b=*/false, b,
                /*c_mod=*/(short)0, c, /*reuse_a=*/false, /*reuse_b=*/false);

        // D layout: VGPR r -> M = r + 8*kh, N = ml
#pragma unroll
        for (int r = 0; r < 8; ++r) {
            const int row = mb + kh * 8 + r;
            dst[row * 33 + hb + ml] = c[r];
        }
    }
    __syncthreads();

    // ---- preload the tiny vectors (uniform -> scalarized by compiler) ----
    float w2r[HH], b1r[HH];
#pragma unroll
    for (int h = 0; h < HH; ++h) { w2r[h] = W2[h]; b1r[h] = b1[h]; }

    // ---- scores[n,i,j] = sum_h W2[h]*relu(P[i,h]+Q[j,h]+b1[h]); b2 cancels
    //      later in the antisymmetrization, so it is omitted here. ----
#pragma unroll 1
    for (int q = 0; q < 64; ++q) {
        const int p = q * 256 + tid;      // consecutive lanes -> consecutive j
        const int i = p >> 7;
        const int j = p & 127;
        float s = 0.f;
#pragma unroll
        for (int h = 0; h < HH; ++h) {
            const float v = Pl[i * 33 + h] + Ql[j * 33 + h] + b1r[h];
            s = fmaf(w2r[h], fmaxf(v, 0.f), s);
        }
        unsafeAtomicAdd(&S[p], s);        // hw global_atomic_add_f32
    }
}

// -----------------------------------------------------------------------------
// Finalize: antisym mean, sigmoids, masks. 16384 threads.
//   antisym_logits = (S - S^T)/NTOK   (b2 and symmetric parts cancel)
// -----------------------------------------------------------------------------
__global__ __launch_bounds__(256)
void cgl_finalize(const float* __restrict__ S,
                  const float* __restrict__ Wmag,
                  float* __restrict__ out)   // [A (128x128) | Wm (128x128)]
{
    const int p = blockIdx.x * blockDim.x + threadIdx.x;
    if (p >= DD * DD) return;
    const int i = p >> 7;
    const int j = p & 127;
    const float offd = (i == j) ? 0.f : 1.f;

    const float al  = (S[i * DD + j] - S[j * DD + i]) * (1.0f / NTOK); // TAU=1
    const float dir = (1.f / (1.f + __expf(-al))) * offd;
    const float wm  = 0.5f * (Wmag[i * DD + j] + Wmag[j * DD + i]) * offd;
    const float A   = (1.f / (1.f + __expf(-wm))) * dir * offd;

    out[p]           = A;    // output 0: A
    out[DD * DD + p] = wm;   // output 1: Wm
}

extern "C" void kernel_launch(void* const* d_in, const int* in_sizes, int n_in,
                              void* d_out, int out_size, void* d_ws, size_t ws_size,
                              hipStream_t stream) {
    const float* z    = (const float*)d_in[0];  // (4,64,128,32)
    const float* Wmag = (const float*)d_in[1];  // (128,128)
    const float* W1   = (const float*)d_in[2];  // (32,64)
    const float* b1   = (const float*)d_in[3];  // (32)
    const float* W2   = (const float*)d_in[4];  // (1,32)
    // d_in[5] = b2 : cancels in the antisymmetrization, unused.

    float* S = (float*)d_ws;                    // 128*128*4 = 64 KB scratch
    hipMemsetAsync(S, 0, DD * DD * sizeof(float), stream);

    cgl_main<<<NTOK, 256, 0, stream>>>(z, W1, b1, W2, S);
    cgl_finalize<<<(DD * DD) / 256, 256, 0, stream>>>(S, Wmag, (float*)d_out);
}
